// CausalAttention_1477468749912
// MI455X (gfx1250) — matrix operand order repaired and checked by reading, not compiled, not run on hardware
//
#include <hip/hip_runtime.h>

// ---------------------------------------------------------------------------
// Causal attention forward for MI455X (gfx1250), wave32 + WMMA bf16.
//   B=4, S=2048, D=1024, f32 in / f32 out.
// Pipeline:
//   K1: Q/K/V = x @ W^T  (f32 -> bf16 in LDS, v_wmma_f32_16x16x32_bf16)
//   K2: S = Q K^T * 1/sqrt(d), causal mask, row softmax -> P (bf16)
//   K3: O = P @ V (causal-bounded K loop)
// ---------------------------------------------------------------------------

typedef __attribute__((ext_vector_type(16))) __bf16        v16bf;
typedef __attribute__((ext_vector_type(8)))  float         v8f;
typedef __attribute__((ext_vector_type(8)))  unsigned int  v8u;

#define DM   1024   // model dim (d_in == d_out)
#define SEQ  2048
#define NB   4

__device__ __forceinline__ unsigned short f32_to_bf16(float f) {
  unsigned int u = __builtin_bit_cast(unsigned int, f);
  u += 0x7FFFu + ((u >> 16) & 1u);          // round-to-nearest-even
  return (unsigned short)(u >> 16);
}

// Packed pair conversion. Prefer single-instruction v_cvt_pk_bf16_f32 when the
// toolchain exposes it; else round-half-up + one v_perm_b32 byte-pack
// (3 VALU per pair vs ~6 for the full RNE sequence).
__device__ __forceinline__ unsigned int pack2_bf16(float a, float b) {
#if __has_builtin(__builtin_amdgcn_cvt_pk_bf16_f32)
  auto p = __builtin_amdgcn_cvt_pk_bf16_f32(a, b);
  return __builtin_bit_cast(unsigned int, p);
#else
  const unsigned int ua = __builtin_bit_cast(unsigned int, a) + 0x8000u;
  const unsigned int ub = __builtin_bit_cast(unsigned int, b) + 0x8000u;
  // dst bytes: [ua.b2, ua.b3, ub.b2, ub.b3]  (src1=ua -> idx 0-3, src0=ub -> 4-7)
  return __builtin_amdgcn_perm(ub, ua, 0x07060302u);
#endif
}

// A fragment 16x32 bf16 (ISA 7.12.2): lane L -> row M = L&15;
// half h -> k = h + (h>=8 ? 8 : 0) + (L>=16 ? 8 : 0). Pairs are contiguous.
__device__ __forceinline__ v16bf load_frag_a(const unsigned short* lds, int stride, int lane) {
  const int m  = lane & 15;
  const int hi = lane >> 4;
  v8u u;
#pragma unroll
  for (int w = 0; w < 8; ++w) {
    const int k = 2 * w + ((w >= 4) ? 8 : 0) + hi * 8;
    u[w] = *(const unsigned int*)(lds + m * stride + k);
  }
  return __builtin_bit_cast(v16bf, u);
}

// B fragment 32x16 bf16: lane L -> col N = L&15; half h -> k = h + (h>=8 ? 8 : 0) + (L>=16 ? 8 : 0).
// LDS tile is stored [n][k] (contiguous in k), so pairs are contiguous dwords.
__device__ __forceinline__ v16bf load_frag_b(const unsigned short* lds, int stride, int lane) {
  const int n  = lane & 15;
  const int hi = lane >> 4;
  v8u u;
#pragma unroll
  for (int w = 0; w < 8; ++w) {
    const int k = 2 * w + ((w >= 4) ? 8 : 0) + hi * 8;
    u[w] = *(const unsigned int*)(lds + n * stride + k);
  }
  return __builtin_bit_cast(v16bf, u);
}

__device__ __forceinline__ v8f wmma_bf16(v16bf a, v16bf b, v8f c) {
  return __builtin_amdgcn_wmma_f32_16x16x32_bf16(false, a, false, b, (short)0, c, false, false);
}

// ---------------------------------------------------------------------------
// K1: QKV projection.  out[row][e] = sum_d x[row][d] * W[e][d]
// Tile M=128 x N=64, 8 waves (4x2), K staged 64 deep -> 8 WMMAs per barrier.
// Staging: float4 global loads, packed bf16 pair (uint2) LDS stores.
// ---------------------------------------------------------------------------
#define K1_AS 72   // LDS row stride (halves), multiple of 4
__global__ __launch_bounds__(256) void qkv_proj_kernel(
    const float* __restrict__ x,
    const float* __restrict__ Wk, const float* __restrict__ Wq, const float* __restrict__ Wv,
    unsigned short* __restrict__ qb, unsigned short* __restrict__ kb,
    unsigned short* __restrict__ vb) {
  extern __shared__ unsigned short sm1[];
  unsigned short* At = sm1;                 // [128][K1_AS] bf16 (x tile)
  unsigned short* Bt = sm1 + 128 * K1_AS;   // [ 64][K1_AS] bf16 (W tile)

  const int tid  = threadIdx.x;
  const int wave = tid >> 5, lane = tid & 31;
  const int mi = wave & 3, ni = wave >> 2;
  const int n0 = blockIdx.x * 64;
  const int m0 = blockIdx.y * 128;          // flattened B*S row

  const float* W;
  unsigned short* outp;
  if (blockIdx.z == 0)      { W = Wk; outp = qb; }   // q = x @ Wk^T (reference swap)
  else if (blockIdx.z == 1) { W = Wq; outp = kb; }   // k = x @ Wq^T
  else                      { W = Wv; outp = vb; }

  v8f acc[2][2];
#pragma unroll
  for (int i = 0; i < 2; ++i)
#pragma unroll
    for (int j = 0; j < 2; ++j) acc[i][j] = (v8f){0.f,0.f,0.f,0.f,0.f,0.f,0.f,0.f};

  for (int dc = 0; dc < DM; dc += 64) {
    // A: 128x64 f32 -> bf16, float4 loads (8192 elems / 4 = 2048 groups)
#pragma unroll
    for (int e = tid; e < 2048; e += 256) {
      const int r = e >> 4, c4 = (e & 15) * 4;
      const float4 f = *(const float4*)(x + (size_t)(m0 + r) * DM + dc + c4);
      uint2 p; p.x = pack2_bf16(f.x, f.y); p.y = pack2_bf16(f.z, f.w);
      *(uint2*)(At + r * K1_AS + c4) = p;
    }
    // B: 64x64 f32 -> bf16 (4096 elems / 4 = 1024 groups)
#pragma unroll
    for (int e = tid; e < 1024; e += 256) {
      const int r = e >> 4, c4 = (e & 15) * 4;
      const float4 f = *(const float4*)(W + (size_t)(n0 + r) * DM + dc + c4);
      uint2 p; p.x = pack2_bf16(f.x, f.y); p.y = pack2_bf16(f.z, f.w);
      *(uint2*)(Bt + r * K1_AS + c4) = p;
    }
    __syncthreads();
#pragma unroll
    for (int dd = 0; dd < 2; ++dd) {
      const v16bf a0 = load_frag_a(At + (mi * 32     ) * K1_AS + dd * 32, K1_AS, lane);
      const v16bf a1 = load_frag_a(At + (mi * 32 + 16) * K1_AS + dd * 32, K1_AS, lane);
      const v16bf b0 = load_frag_b(Bt + (ni * 32     ) * K1_AS + dd * 32, K1_AS, lane);
      const v16bf b1 = load_frag_b(Bt + (ni * 32 + 16) * K1_AS + dd * 32, K1_AS, lane);
      acc[0][0] = wmma_bf16(a0, b0, acc[0][0]);
      acc[0][1] = wmma_bf16(a0, b1, acc[0][1]);
      acc[1][0] = wmma_bf16(a1, b0, acc[1][0]);
      acc[1][1] = wmma_bf16(a1, b1, acc[1][1]);
    }
    __syncthreads();
  }

  const int nlo = lane & 15, mhi = (lane >> 4) * 8;
#pragma unroll
  for (int im = 0; im < 2; ++im)
#pragma unroll
    for (int in = 0; in < 2; ++in)
#pragma unroll
      for (int r = 0; r < 8; ++r) {
        const int row = m0 + mi * 32 + im * 16 + mhi + r;
        const int col = n0 + ni * 32 + in * 16 + nlo;
        outp[(size_t)row * DM + col] = f32_to_bf16(acc[im][in][r]);
      }
}

// ---------------------------------------------------------------------------
// K2: scores + causal softmax -> P (bf16). One workgroup = 16 query rows;
// full 16x2048 f32 score strip in LDS (CDNA5: 320KB/WGP). 8 waves tile 128
// k-cols per pass; K staged 128 d-columns at a time -> 4 WMMAs per barrier.
// ---------------------------------------------------------------------------
#define K2_QS 1032  // Qt row stride (halves)
#define K2_KS 136   // Kt row stride (halves), multiple of 8 for uint4 stores
__global__ __launch_bounds__(256) void attn_scores_kernel(
    const unsigned short* __restrict__ qb, const unsigned short* __restrict__ kb,
    unsigned short* __restrict__ P) {
  extern __shared__ unsigned short sm2[];
  unsigned short* Qt = sm2;                                        // [16][K2_QS]
  unsigned short* Kt = sm2 + 16 * K2_QS;                           // [128][K2_KS]
  float* Stile = (float*)(sm2 + 16 * K2_QS + 128 * K2_KS);         // [16][2048]

  const int tid = threadIdx.x, wave = tid >> 5, lane = tid & 31;
  const int b  = blockIdx.y;
  const int q0 = blockIdx.x * 16;
  // k-extent rounded so every 64-aligned consumer tile in K3 reads only
  // blocks we fully write here.
  const int keb = (((q0 & ~63) + 64) + 127) / 128;
  const int KE  = keb * 128;

  // Stage Q strip once: 16x1024 bf16, uint4 loads/stores (2048 vec4 groups).
#pragma unroll
  for (int e = tid; e < 2048; e += 256) {
    const int r = e >> 7, c8 = (e & 127) * 8;
    *(uint4*)(Qt + r * K2_QS + c8) =
        *(const uint4*)(qb + (size_t)(b * SEQ + q0 + r) * DM + c8);
  }
  __syncthreads();

  const float scale = 0.03125f;   // 1/sqrt(1024)
  for (int kbk = 0; kbk < keb; ++kbk) {
    const int kbase = kbk * 128;
    v8f acc = (v8f){0.f,0.f,0.f,0.f,0.f,0.f,0.f,0.f};
    for (int dc = 0; dc < DM; dc += 128) {
      // Stage Kt[128][128]: 16384 halves = 2048 uint4 groups (8 per thread).
#pragma unroll
      for (int e = tid; e < 2048; e += 256) {
        const int r = e >> 4, c8 = (e & 15) * 8;
        const unsigned short* src =
            kb + (size_t)(b * SEQ + kbase + r) * DM + dc + c8;
        __builtin_prefetch(src + 128, 0, 1);     // next d-chunk -> global_prefetch_b8
        *(uint4*)(Kt + r * K2_KS + c8) = *(const uint4*)src;
      }
      __syncthreads();
#pragma unroll
      for (int dd = 0; dd < 4; ++dd) {
        const v16bf a  = load_frag_a(Qt + dc + dd * 32, K2_QS, lane);
        const v16bf bf = load_frag_b(Kt + (wave * 16) * K2_KS + dd * 32, K2_KS, lane);
        acc = wmma_bf16(a, bf, acc);
      }
      __syncthreads();
    }
    const int nlo = lane & 15, mhi = (lane >> 4) * 8;
#pragma unroll
    for (int r = 0; r < 8; ++r) {
      const int m    = mhi + r;
      const int kcol = kbase + wave * 16 + nlo;
      const float sc = (kcol <= q0 + m) ? acc[r] * scale : -__builtin_inff();
      Stile[m * 2048 + kcol] = sc;
    }
  }
  __syncthreads();

  // Row softmax: wave w owns rows {w, w+8}; 32-lane strided + shfl_xor reduce.
#pragma unroll
  for (int rr = 0; rr < 2; ++rr) {
    const int m = wave + rr * 8;
    float mx = -__builtin_inff();
    for (int c = lane; c < KE; c += 32) mx = fmaxf(mx, Stile[m * 2048 + c]);
#pragma unroll
    for (int off = 16; off > 0; off >>= 1) mx = fmaxf(mx, __shfl_xor(mx, off, 32));
    float sum = 0.f;
    for (int c = lane; c < KE; c += 32) {
      const float p = __expf(Stile[m * 2048 + c] - mx);   // exp(-inf)=0 for masked
      Stile[m * 2048 + c] = p;
      sum += p;
    }
#pragma unroll
    for (int off = 16; off > 0; off >>= 1) sum += __shfl_xor(sum, off, 32);
    const float inv = 1.0f / sum;
    // Packed write-out: each lane owns two adjacent columns -> one b32 store.
    for (int c = lane * 2; c < KE; c += 64) {
      const unsigned int pk = pack2_bf16(Stile[m * 2048 + c] * inv,
                                         Stile[m * 2048 + c + 1] * inv);
      *(unsigned int*)(P + ((size_t)b * SEQ + q0 + m) * SEQ + c) = pk;
    }
  }
}

// ---------------------------------------------------------------------------
// K3: O = P @ V, causal-bounded K loop. Tile M=64 x N=128, 8 waves (2x4).
// K staged 64 deep -> 8 WMMAs per barrier pair. V staged transposed into LDS
// so B fragments read contiguous k.
// ---------------------------------------------------------------------------
#define K3_S 72   // LDS row stride (halves), multiple of 8
__global__ __launch_bounds__(256) void attn_av_kernel(
    const unsigned short* __restrict__ P, const unsigned short* __restrict__ vb,
    float* __restrict__ out) {
  extern __shared__ unsigned short sm3[];
  unsigned short* At = sm3;                // [ 64][K3_S] bf16 (P tile)
  unsigned short* Bt = sm3 + 64 * K3_S;    // [128][K3_S] bf16 (V tile, transposed [e][k])

  const int tid = threadIdx.x, wave = tid >> 5, lane = tid & 31;
  const int mi = wave & 1, ni = wave >> 1;
  const int b  = blockIdx.z;
  const int e0 = blockIdx.x * 128;
  const int Q0 = blockIdx.y * 64;
  const int KE = ((Q0 + 64 + 127) / 128) * 128;   // matches K2 write extent

  v8f acc[2][2];
#pragma unroll
  for (int i = 0; i < 2; ++i)
#pragma unroll
    for (int j = 0; j < 2; ++j) acc[i][j] = (v8f){0.f,0.f,0.f,0.f,0.f,0.f,0.f,0.f};

  for (int kc = 0; kc < KE; kc += 64) {
    // A: P tile 64x64 bf16 = 512 uint4 groups (2 per thread).
#pragma unroll
    for (int e = tid; e < 512; e += 256) {
      const int r = e >> 3, c8 = (e & 7) * 8;
      const unsigned short* src = P + ((size_t)b * SEQ + Q0 + r) * SEQ + kc + c8;
      __builtin_prefetch(src + 64, 0, 1);          // next k-chunk
      *(uint4*)(At + r * K3_S + c8) = *(const uint4*)src;
    }
    // B: V tile 64(k) x 128(e), read uint4 along e, scatter transposed.
#pragma unroll
    for (int e = tid; e < 1024; e += 256) {
      const int kk = e >> 4, e8 = (e & 15) * 8;
      const uint4 u = *(const uint4*)(vb + (size_t)(b * SEQ + kc + kk) * DM + e0 + e8);
      unsigned short t[8];
      *(uint4*)t = u;
#pragma unroll
      for (int j = 0; j < 8; ++j) Bt[(e8 + j) * K3_S + kk] = t[j];
    }
    __syncthreads();
#pragma unroll
    for (int dd = 0; dd < 2; ++dd) {
      const v16bf a0 = load_frag_a(At + (mi * 32     ) * K3_S + dd * 32, K3_S, lane);
      const v16bf a1 = load_frag_a(At + (mi * 32 + 16) * K3_S + dd * 32, K3_S, lane);
      const v16bf b0 = load_frag_b(Bt + (ni * 32     ) * K3_S + dd * 32, K3_S, lane);
      const v16bf b1 = load_frag_b(Bt + (ni * 32 + 16) * K3_S + dd * 32, K3_S, lane);
      acc[0][0] = wmma_bf16(a0, b0, acc[0][0]);
      acc[0][1] = wmma_bf16(a0, b1, acc[0][1]);
      acc[1][0] = wmma_bf16(a1, b0, acc[1][0]);
      acc[1][1] = wmma_bf16(a1, b1, acc[1][1]);
    }
    __syncthreads();
  }

  const int nlo = lane & 15, mhi = (lane >> 4) * 8;
#pragma unroll
  for (int im = 0; im < 2; ++im)
#pragma unroll
    for (int in = 0; in < 2; ++in)
#pragma unroll
      for (int r = 0; r < 8; ++r) {
        const int row = Q0 + mi * 32 + im * 16 + mhi + r;
        const int col = e0 + ni * 32 + in * 16 + nlo;
        out[((size_t)b * SEQ + row) * DM + col] = acc[im][in][r];
      }
}

// ---------------------------------------------------------------------------
extern "C" void kernel_launch(void* const* d_in, const int* in_sizes, int n_in,
                              void* d_out, int out_size, void* d_ws, size_t ws_size,
                              hipStream_t stream) {
  const float* x  = (const float*)d_in[0];
  const float* Wk = (const float*)d_in[1];
  const float* Wq = (const float*)d_in[2];
  const float* Wv = (const float*)d_in[3];

  // Workspace layout (80 MB total): qb/kb/vb bf16 [B*S][D], P bf16 [B][S][S].
  unsigned short* qb = (unsigned short*)d_ws;
  unsigned short* kb = qb + (size_t)NB * SEQ * DM;
  unsigned short* vb = kb + (size_t)NB * SEQ * DM;
  unsigned short* P  = vb + (size_t)NB * SEQ * DM;
  float* out = (float*)d_out;

  const size_t sm1 = (size_t)(128 * K1_AS + 64 * K1_AS) * sizeof(unsigned short);
  const size_t sm2 = (size_t)(16 * K2_QS + 128 * K2_KS) * sizeof(unsigned short)
                   + (size_t)16 * 2048 * sizeof(float);   // ~199 KB (< 320 KB WGP LDS)
  const size_t sm3 = (size_t)(64 * K3_S + 128 * K3_S) * sizeof(unsigned short);

  qkv_proj_kernel<<<dim3(DM / 64, (NB * SEQ) / 128, 3), 256, sm1, stream>>>(
      x, Wk, Wq, Wv, qb, kb, vb);
  attn_scores_kernel<<<dim3(SEQ / 16, NB), 256, sm2, stream>>>(qb, kb, P);
  attn_av_kernel<<<dim3(DM / 128, SEQ / 64, NB), 256, sm3, stream>>>(P, vb, out);
}
